// PAP2PNet_52012053954783
// MI455X (gfx1250) — matrix operand order repaired and checked by reading, not compiled, not run on hardware
//
#include <hip/hip_runtime.h>
#include <hip/hip_bf16.h>

#define HIDDEN 256
#define LEVELS 4

typedef __attribute__((ext_vector_type(16))) __bf16 bf16x16;
typedef __attribute__((ext_vector_type(8)))  float  f32x8;

union Frag { bf16x16 v; unsigned u[8]; };

// ---------------------------------------------------------------------------
// Weight convert + pad: src fp32 [outReal][K] row-major -> dst bf16 [outPad][K]
// (rows >= outReal zero-filled).  Row-major [out][in] is exactly the B^T
// fragment layout WMMA wants: consecutive K pairs per dword per output column.
// ---------------------------------------------------------------------------
__global__ void k_cvt_pad(const float* __restrict__ src, __bf16* __restrict__ dst,
                          int outReal, int K, int outPad) {
    int i = blockIdx.x * blockDim.x + threadIdx.x;
    int total = outPad * K;
    if (i >= total) return;
    int o = i / K;
    dst[i] = (o < outReal) ? (__bf16)src[i] : (__bf16)0.0f;
}

// ---------------------------------------------------------------------------
// Bilinear sample one point, 256 channels spread over the 32 lanes of a wave.
// feat layout: [B][C][H][W] fp32; zero padding outside, align_corners=True.
// ---------------------------------------------------------------------------
__device__ __forceinline__ void bilinear_row(const float* __restrict__ feat,
                                             int b, int H, int W,
                                             float px, float py,
                                             __bf16* __restrict__ dstRow, int lane) {
    const float ix = (px + 1.0f) * 0.5f * (float)(W - 1);
    const float iy = (py + 1.0f) * 0.5f * (float)(H - 1);
    const float x0f = floorf(ix), y0f = floorf(iy);
    const int x0 = (int)x0f, y0 = (int)y0f;
    const float wx1 = ix - x0f, wy1 = iy - y0f;
    const float w00 = (1.0f - wx1) * (1.0f - wy1);
    const float w01 = wx1 * (1.0f - wy1);
    const float w10 = (1.0f - wx1) * wy1;
    const float w11 = wx1 * wy1;

    int  idx[4]; float wgt[4];
    const int xs[4] = {x0, x0 + 1, x0,     x0 + 1};
    const int ys[4] = {y0, y0,     y0 + 1, y0 + 1};
    const float ws4[4] = {w00, w01, w10, w11};
#pragma unroll
    for (int q = 0; q < 4; ++q) {
        bool v = (xs[q] >= 0) && (xs[q] <= W - 1) && (ys[q] >= 0) && (ys[q] <= H - 1);
        int xc = min(max(xs[q], 0), W - 1);
        int yc = min(max(ys[q], 0), H - 1);
        idx[q] = yc * W + xc;
        wgt[q] = v ? ws4[q] : 0.0f;
    }
    const size_t HW   = (size_t)H * (size_t)W;
    const float* base = feat + (size_t)b * HIDDEN * HW;
#pragma unroll 2
    for (int c = lane; c < HIDDEN; c += 32) {
        const float* ch = base + (size_t)c * HW;
        float acc = wgt[0] * ch[idx[0]] + wgt[1] * ch[idx[1]]
                  + wgt[2] * ch[idx[2]] + wgt[3] * ch[idx[3]];
        dstRow[c] = (__bf16)acc;
    }
}

// ---------------------------------------------------------------------------
// 16-row GEMM layer on one wave:
//   Y[16][Nout] = act( X[16][K](bf16, LDS) @ W[Nout][K](bf16, global)^T + b )
// WMMA bf16 16x16x32; fragment layouts per CDNA5 ISA 7.12.2.
// ---------------------------------------------------------------------------
__device__ __forceinline__ void wave_gemm(const __bf16* __restrict__ Xl, int K,
                                          const __bf16* __restrict__ Wg,
                                          const float* __restrict__ bias, int Nout,
                                          __bf16* __restrict__ Yl, int ldY,
                                          bool relu, int lane) {
    const int h = lane >> 4;     // lane half
    const int t = lane & 15;     // column / row-in-half index
    for (int n0 = 0; n0 < Nout; n0 += 16) {
        __builtin_prefetch(Wg + (size_t)(n0 + t) * K, 0, 1);
        f32x8 acc = {};
        for (int k0 = 0; k0 < K; k0 += 32) {
            Frag A, B;
#pragma unroll
            for (int j = 0; j < 8; ++j) {
                // A 16x32 bf16: lane half h, row t; VGPR j holds K pair:
                int ka = ((j & 4) ? 16 : 0) + h * 8 + 2 * (j & 3);
                A.u[j] = *(const unsigned*)(Xl + (size_t)t * K + (k0 + ka));
                // B 32x16 bf16: column t; VGPR j holds same K pair as A:
                int kb = ((j & 4) ? 16 : 0) + h * 8 + 2 * (j & 3);
                B.u[j] = *(const unsigned*)(Wg + (size_t)(n0 + t) * K + (k0 + kb));
            }
            acc = __builtin_amdgcn_wmma_f32_16x16x32_bf16(
                false, A.v, false, B.v, (short)0, acc, false, false);
        }
        const float bv = bias ? bias[n0 + t] : 0.0f;
#pragma unroll
        for (int j = 0; j < 8; ++j) {              // C/D: row M = 8h + j, col t
            float y = acc[j] + bv;
            if (relu) y = fmaxf(y, 0.0f);
            Yl[(size_t)(h * 8 + j) * ldY + (n0 + t)] = (__bf16)y;
        }
    }
}

// Single 16x16 output tile (final layers padded to 16 rows), result -> LDS fp32.
__device__ __forceinline__ void wave_gemm_tile16(const __bf16* __restrict__ Xl, int K,
                                                 const __bf16* __restrict__ Wp,
                                                 float* __restrict__ tileOut, int lane) {
    const int h = lane >> 4, t = lane & 15;
    f32x8 acc = {};
    for (int k0 = 0; k0 < K; k0 += 32) {
        Frag A, B;
#pragma unroll
        for (int j = 0; j < 8; ++j) {
            int ka = ((j & 4) ? 16 : 0) + h * 8 + 2 * (j & 3);
            A.u[j] = *(const unsigned*)(Xl + (size_t)t * K + (k0 + ka));
            int kb = ((j & 4) ? 16 : 0) + h * 8 + 2 * (j & 3);
            B.u[j] = *(const unsigned*)(Wp + (size_t)t * K + (k0 + kb));
        }
        acc = __builtin_amdgcn_wmma_f32_16x16x32_bf16(
            false, A.v, false, B.v, (short)0, acc, false, false);
    }
#pragma unroll
    for (int j = 0; j < 8; ++j)
        tileOut[(h * 8 + j) * 16 + t] = acc[j];
}

// ---------------------------------------------------------------------------
// Stage 0: sample level-0 at anchors, deform MLP, write deformed points.
// One wave (32 threads) per 16-row tile.
// ---------------------------------------------------------------------------
__global__ void __launch_bounds__(32)
k_deform(const float* __restrict__ feat0, const float* __restrict__ points,
         const __bf16* __restrict__ dw1b, const float* __restrict__ db1,
         const __bf16* __restrict__ dw2p, const float* __restrict__ db2,
         float* __restrict__ ptsOut, int NN, int H0, int W0, float stride0) {
    __shared__ alignas(16) __bf16 preB[16 * HIDDEN];
    __shared__ alignas(16) __bf16 hidB[16 * HIDDEN];
    __shared__ alignas(16) float  tileOut[16 * 16];
    __shared__ float ptxy[16][2];

    const int lane = threadIdx.x;
    const int r0   = blockIdx.x * 16;

    {   // load 16 points (32 scalars) with the full wave
        int r = lane >> 1, cmp = lane & 1;
        ptxy[r][cmp] = points[(size_t)(r0 + r) * 2 + cmp];
    }
    __syncthreads();

    for (int r = 0; r < 16; ++r) {
        const int g = r0 + r;
        const int b = g / NN;
        const float px = 2.0f * ptxy[r][0] / (stride0 * (float)W0) - 1.0f;
        const float py = 2.0f * ptxy[r][1] / (stride0 * (float)H0) - 1.0f;
        bilinear_row(feat0, b, H0, W0, px, py, preB + (size_t)r * HIDDEN, lane);
    }
    __syncthreads();

    wave_gemm(preB, HIDDEN, dw1b, db1, HIDDEN, hidB, HIDDEN, true, lane);
    __syncthreads();
    wave_gemm_tile16(hidB, HIDDEN, dw2p, tileOut, lane);
    __syncthreads();

    {   // offsets = cols 0..1 (+ bias db2[cmp] added here)
        int r = lane >> 1, cmp = lane & 1;
        float off = tileOut[r * 16 + cmp] + db2[cmp];
        ptsOut[(size_t)(r0 + r) * 2 + cmp] = ptxy[r][cmp] + off;
    }
}

// ---------------------------------------------------------------------------
// Main stage: 4-level sampling, attention aggregation, reg/cls heads.
// One wave per 16-row tile.  LDS ~50 KB.
// ---------------------------------------------------------------------------
__global__ void __launch_bounds__(32)
k_main(const float* __restrict__ f0, const float* __restrict__ f1,
       const float* __restrict__ f2, const float* __restrict__ f3,
       const float* __restrict__ ptsIn,
       const __bf16* __restrict__ lw1b, const float* __restrict__ lb1,
       const __bf16* __restrict__ lw2p, const float* __restrict__ lb2,
       const __bf16* __restrict__ aw1b, const float* __restrict__ ab1,
       const __bf16* __restrict__ aw2p, const float* __restrict__ ab2,
       const __bf16* __restrict__ rw1b, const float* __restrict__ rb1,
       const __bf16* __restrict__ rw2p, const float* __restrict__ rb2,
       const __bf16* __restrict__ cw1b, const float* __restrict__ cb1,
       const __bf16* __restrict__ cw2p, const float* __restrict__ cb2,
       float* __restrict__ outCoords, float* __restrict__ outLogits,
       int NN, int Hbase) {
    __shared__ alignas(16) __bf16 roiB[16 * (LEVELS * HIDDEN)];  // 32 KB
    __shared__ alignas(16) __bf16 hidB[16 * HIDDEN];             //  8 KB
    __shared__ alignas(16) __bf16 featB[16 * HIDDEN];            //  8 KB
    __shared__ alignas(16) float  tileOut[16 * 16];
    __shared__ float attnR[16][LEVELS];
    __shared__ float attnC[16][LEVELS];
    __shared__ float ptxy[16][2];

    const int lane = threadIdx.x;
    const int r0   = blockIdx.x * 16;
    const int KBIG = LEVELS * HIDDEN;

    {
        int r = lane >> 1, cmp = lane & 1;
        ptxy[r][cmp] = ptsIn[(size_t)(r0 + r) * 2 + cmp];
    }
    __syncthreads();

    const float* feats[LEVELS] = {f0, f1, f2, f3};
#pragma unroll
    for (int l = 0; l < LEVELS; ++l) {
        const int   Hl = Hbase >> l;            // 256,128,64,32
        const float sl = (float)(4 << l);       // 4,8,16,32
        for (int r = 0; r < 16; ++r) {
            const int g = r0 + r;
            const int b = g / NN;
            const float px = 2.0f * ptxy[r][0] / (sl * (float)Hl) - 1.0f;
            const float py = 2.0f * ptxy[r][1] / (sl * (float)Hl) - 1.0f;
            bilinear_row(feats[l], b, Hl, Hl, px, py,
                         roiB + (size_t)r * KBIG + l * HIDDEN, lane);
        }
    }
    __syncthreads();

    // ---- reg attention: K=1024 -> 256 (ReLU) -> 4 -> softmax
    wave_gemm(roiB, KBIG, lw1b, lb1, HIDDEN, hidB, HIDDEN, true, lane);
    __syncthreads();
    wave_gemm_tile16(hidB, HIDDEN, lw2p, tileOut, lane);
    __syncthreads();
    if (lane < 16) {
        float v[LEVELS], m = -1e30f, s = 0.0f;
#pragma unroll
        for (int l = 0; l < LEVELS; ++l) { v[l] = tileOut[lane * 16 + l] + lb2[l]; m = fmaxf(m, v[l]); }
#pragma unroll
        for (int l = 0; l < LEVELS; ++l) { v[l] = __expf(v[l] - m); s += v[l]; }
#pragma unroll
        for (int l = 0; l < LEVELS; ++l) attnR[lane][l] = v[l] / s;
    }
    __syncthreads();

    // ---- cls attention
    wave_gemm(roiB, KBIG, aw1b, ab1, HIDDEN, hidB, HIDDEN, true, lane);
    __syncthreads();
    wave_gemm_tile16(hidB, HIDDEN, aw2p, tileOut, lane);
    __syncthreads();
    if (lane < 16) {
        float v[LEVELS], m = -1e30f, s = 0.0f;
#pragma unroll
        for (int l = 0; l < LEVELS; ++l) { v[l] = tileOut[lane * 16 + l] + ab2[l]; m = fmaxf(m, v[l]); }
#pragma unroll
        for (int l = 0; l < LEVELS; ++l) { v[l] = __expf(v[l] - m); s += v[l]; }
#pragma unroll
        for (int l = 0; l < LEVELS; ++l) attnC[lane][l] = v[l] / s;
    }
    __syncthreads();

    // ---- reg branch: weighted level sum -> head MLP -> coords (+pts)
    for (int r = 0; r < 16; ++r) {
        const float a0 = attnR[r][0], a1 = attnR[r][1], a2 = attnR[r][2], a3 = attnR[r][3];
        const __bf16* row = roiB + (size_t)r * KBIG;
        for (int c = lane; c < HIDDEN; c += 32) {
            float s = a0 * (float)row[c] + a1 * (float)row[HIDDEN + c]
                    + a2 * (float)row[2 * HIDDEN + c] + a3 * (float)row[3 * HIDDEN + c];
            featB[(size_t)r * HIDDEN + c] = (__bf16)s;
        }
    }
    __syncthreads();
    wave_gemm(featB, HIDDEN, rw1b, rb1, HIDDEN, hidB, HIDDEN, true, lane);
    __syncthreads();
    wave_gemm_tile16(hidB, HIDDEN, rw2p, tileOut, lane);
    __syncthreads();
    {
        int r = lane >> 1, cmp = lane & 1;
        outCoords[(size_t)(r0 + r) * 2 + cmp] =
            tileOut[r * 16 + cmp] + rb2[cmp] + ptxy[r][cmp];
    }
    __syncthreads();

    // ---- cls branch
    for (int r = 0; r < 16; ++r) {
        const float a0 = attnC[r][0], a1 = attnC[r][1], a2 = attnC[r][2], a3 = attnC[r][3];
        const __bf16* row = roiB + (size_t)r * KBIG;
        for (int c = lane; c < HIDDEN; c += 32) {
            float s = a0 * (float)row[c] + a1 * (float)row[HIDDEN + c]
                    + a2 * (float)row[2 * HIDDEN + c] + a3 * (float)row[3 * HIDDEN + c];
            featB[(size_t)r * HIDDEN + c] = (__bf16)s;
        }
    }
    __syncthreads();
    wave_gemm(featB, HIDDEN, cw1b, cb1, HIDDEN, hidB, HIDDEN, true, lane);
    __syncthreads();
    wave_gemm_tile16(hidB, HIDDEN, cw2p, tileOut, lane);
    __syncthreads();
    {
        int r = lane >> 1, cmp = lane & 1;
        outLogits[(size_t)(r0 + r) * 2 + cmp] = tileOut[r * 16 + cmp] + cb2[cmp];
    }
}

// ---------------------------------------------------------------------------
// Host side
// ---------------------------------------------------------------------------
extern "C" void kernel_launch(void* const* d_in, const int* in_sizes, int n_in,
                              void* d_out, int out_size, void* d_ws, size_t ws_size,
                              hipStream_t stream) {
    (void)n_in; (void)ws_size;
    const float* feat0  = (const float*)d_in[0];
    const float* feat1  = (const float*)d_in[1];
    const float* feat2  = (const float*)d_in[2];
    const float* feat3  = (const float*)d_in[3];
    const float* points = (const float*)d_in[4];
    const float* dw1 = (const float*)d_in[5],  *db1 = (const float*)d_in[6];
    const float* dw2 = (const float*)d_in[7],  *db2 = (const float*)d_in[8];
    const float* rw1 = (const float*)d_in[9],  *rb1 = (const float*)d_in[10];
    const float* rw2 = (const float*)d_in[11], *rb2 = (const float*)d_in[12];
    const float* cw1 = (const float*)d_in[13], *cb1 = (const float*)d_in[14];
    const float* cw2 = (const float*)d_in[15], *cb2 = (const float*)d_in[16];
    const float* lw1 = (const float*)d_in[17], *lb1 = (const float*)d_in[18];
    const float* lw2 = (const float*)d_in[19], *lb2 = (const float*)d_in[20];
    const float* aw1 = (const float*)d_in[21], *ab1 = (const float*)d_in[22];
    const float* aw2 = (const float*)d_in[23], *ab2 = (const float*)d_in[24];

    const int R  = in_sizes[4] / 2;                       // B*N rows (10240)
    // B from feat0: elements = B * HIDDEN * H0 * W0; derive H0 assuming square
    long long hw0 = 1;
    {   // feat0 HW = in_sizes[0] / (B*HIDDEN); find B first via feat3/feat0 ratio-free path:
        // strides are 4..32 with H=W=base/stride; base^2 = in_sizes[0]*16/(in_sizes[0]/ (in_sizes[0]))... keep simple:
        // assume B = 2 unless sizes say otherwise: feat0 = B*256*H0*H0 and H0 = 8*H3, feat3 = B*256*H3*H3
        hw0 = (long long)in_sizes[0] / HIDDEN;            // = B*H0*W0
    }
    // find integer H0 with B*H0*H0 == hw0 for B in {1,2,4,8}
    int Bsz = 2, H0 = 256;
    for (int bTry = 1; bTry <= 8; bTry <<= 1) {
        long long per = hw0 / bTry;
        long long r = 1; while ((r + 1) * (r + 1) <= per) ++r;
        if (r * r == per && hw0 % bTry == 0) { Bsz = bTry; H0 = (int)r; break; }
    }
    const int NN = R / Bsz;
    const int nTiles = R / 16;

    // workspace layout
    char*  ws     = (char*)d_ws;
    float* ptsWs  = (float*)ws;
    size_t off    = (size_t)R * 2 * sizeof(float);
    auto wsbf = [&](size_t elems) { __bf16* p = (__bf16*)(ws + off); off += elems * sizeof(__bf16); return p; };
    __bf16* dw1b = wsbf(256 * 256);  __bf16* dw2p = wsbf(16 * 256);
    __bf16* rw1b = wsbf(256 * 256);  __bf16* rw2p = wsbf(16 * 256);
    __bf16* cw1b = wsbf(256 * 256);  __bf16* cw2p = wsbf(16 * 256);
    __bf16* lw1b = wsbf(256 * 1024); __bf16* lw2p = wsbf(16 * 256);
    __bf16* aw1b = wsbf(256 * 1024); __bf16* aw2p = wsbf(16 * 256);

    auto cvt = [&](const float* s, __bf16* d, int outReal, int K, int outPad) {
        int total = outPad * K;
        k_cvt_pad<<<(total + 255) / 256, 256, 0, stream>>>(s, d, outReal, K, outPad);
    };
    cvt(dw1, dw1b, 256, 256, 256);   cvt(dw2, dw2p, 2, 256, 16);
    cvt(rw1, rw1b, 256, 256, 256);   cvt(rw2, rw2p, 2, 256, 16);
    cvt(cw1, cw1b, 256, 256, 256);   cvt(cw2, cw2p, 2, 256, 16);
    cvt(lw1, lw1b, 256, 1024, 256);  cvt(lw2, lw2p, 4, 256, 16);
    cvt(aw1, aw1b, 256, 1024, 256);  cvt(aw2, aw2p, 4, 256, 16);

    k_deform<<<nTiles, 32, 0, stream>>>(feat0, points, dw1b, db1, dw2p, db2,
                                        ptsWs, NN, H0, H0, 4.0f);

    float* outCoords = (float*)d_out;
    float* outLogits = outCoords + (size_t)R * 2;
    (void)out_size;
    k_main<<<nTiles, 32, 0, stream>>>(feat0, feat1, feat2, feat3, ptsWs,
                                      lw1b, lb1, lw2p, lb2,
                                      aw1b, ab1, aw2p, ab2,
                                      rw1b, rb1, rw2p, rb2,
                                      cw1b, cb1, cw2p, cb2,
                                      outCoords, outLogits, NN, H0);
}
